// MoELayer_86904368268139
// MI455X (gfx1250) — matrix operand order repaired and checked by reading, not compiled, not run on hardware
//
#include <hip/hip_runtime.h>

#define D_DIM 1024
#define H_DIM 4096
#define E_NUM 8
#define NTOK  8192          // 4 * 2048
#define MTILE 64
#define KC    256
#define LDA   (KC + 8)      // +16B pad per row -> rotates LDS banks
#define MAXTILES 136        // Sum_e ceil((n_e+1)/64) <= (8200 + 8*63)/64 = 136

typedef __attribute__((ext_vector_type(16))) __bf16 v16bf;
typedef __attribute__((ext_vector_type(8)))  float  v8f;
typedef __attribute__((ext_vector_type(4)))  int    v4i;

union FragBF { v16bf v; unsigned short s[16]; unsigned u[8]; uint4 q[2]; };

// f32 -> bf16 (round-half-up; folds into d16_hi stores)
__device__ __forceinline__ unsigned short f2bf_r(float f) {
    unsigned u = __builtin_bit_cast(unsigned, f) + 0x8000u;
    return (unsigned short)(u >> 16);
}

// packed f32x2 -> bf16x2 (low = a, high = b): 2x v_add + 1x v_perm_b32
__device__ __forceinline__ unsigned pk2bf_perm(float a, float b) {
    unsigned ua = __builtin_bit_cast(unsigned, a) + 0x8000u;
    unsigned ub = __builtin_bit_cast(unsigned, b) + 0x8000u;
    return __builtin_amdgcn_perm(ub, ua, 0x07060302u);  // {ub[31:16], ua[31:16]}
}

#if __has_builtin(__builtin_amdgcn_cvt_pk_bf16_f32)
#define PK2BF(a, b) (__builtin_bit_cast(unsigned, __builtin_amdgcn_cvt_pk_bf16_f32((a), (b))))
#else
#define PK2BF(a, b) pk2bf_perm((a), (b))
#endif

// async global->LDS copy path (CDNA5); param types per compiler diagnostic probe
#if __has_builtin(__builtin_amdgcn_global_load_async_to_lds_b128) && \
    __has_builtin(__builtin_amdgcn_s_wait_asynccnt)
#define USE_ASYNC 1
typedef __attribute__((address_space(1))) v4i* gv4i_p;
typedef __attribute__((address_space(3))) v4i* lv4i_p;
#else
#define USE_ASYNC 0
#endif

// ---------------- gate: logits, softmax, argmax, counts ----------------
__global__ __launch_bounds__(256) void k_init(int* counts) {
    if (threadIdx.x < E_NUM) counts[threadIdx.x] = 0;
}

__global__ __launch_bounds__(256) void k_gate(const float* __restrict__ x,
                                              const float* __restrict__ Wg,
                                              const float* __restrict__ bg,
                                              float* __restrict__ probsOut,
                                              int* __restrict__ idxOut,
                                              int* __restrict__ counts) {
    int wave = threadIdx.x >> 5;
    int lane = threadIdx.x & 31;
    int t = blockIdx.x * 8 + wave;                 // one wave32 per token
    const float* xr = x + (size_t)t * D_DIM;

    float acc[E_NUM];
#pragma unroll
    for (int e = 0; e < E_NUM; ++e) acc[e] = 0.f;
    for (int k = 0; k < D_DIM / 32; ++k) {
        int d = k * 32 + lane;                     // coalesced x reads
        float xv = xr[d];
        const float* wr = Wg + (size_t)d * E_NUM;
#pragma unroll
        for (int e = 0; e < E_NUM; ++e) acc[e] = fmaf(xv, wr[e], acc[e]);
    }
#pragma unroll
    for (int e = 0; e < E_NUM; ++e) {
#pragma unroll
        for (int off = 16; off >= 1; off >>= 1)
            acc[e] += __shfl_xor(acc[e], off, 32);
    }
    if (lane == 0) {
        float lg[E_NUM], mx = -3.0e38f;
#pragma unroll
        for (int e = 0; e < E_NUM; ++e) { lg[e] = acc[e] + bg[e]; mx = fmaxf(mx, lg[e]); }
        float s = 0.f;
#pragma unroll
        for (int e = 0; e < E_NUM; ++e) { lg[e] = __expf(lg[e] - mx); s += lg[e]; }
        float inv = 1.0f / s;
        int best = 0; float bv = lg[0];
#pragma unroll
        for (int e = 0; e < E_NUM; ++e) {
            probsOut[(size_t)t * E_NUM + e] = lg[e] * inv;
            if (lg[e] > bv) { bv = lg[e]; best = e; }   // first max, matches argmax
        }
        idxOut[t] = best;
        atomicAdd(&counts[best], 1);
    }
}

// offsets; slot 0 of each expert segment = sentinel (-1) zero-token -> yields C_e
__global__ void k_scan(const int* __restrict__ counts, int* __restrict__ offs,
                       int* __restrict__ cursor, int* __restrict__ perm) {
    if (threadIdx.x == 0) {
        int o = 0; offs[0] = 0;
        for (int e = 0; e < E_NUM; ++e) {
            cursor[e] = o + 1;
            perm[o] = -1;
            o += counts[e] + 1;
            offs[e + 1] = o;
        }
    }
}

__global__ __launch_bounds__(256) void k_scatter(const int* __restrict__ idx,
                                                 int* __restrict__ cursor,
                                                 int* __restrict__ perm) {
    int t = blockIdx.x * 256 + threadIdx.x;
    int e = idx[t];
    int slot = atomicAdd(&cursor[e], 1);
    perm[slot] = t;
}

// map flat tile id -> (expert, tileStart row in perm space, valid rows)
__device__ __forceinline__ bool tileMap(const int* __restrict__ offs, int tileId,
                                        int& e, int& tileStart, int& rowsLeft) {
    int t = tileId;
    int o0 = offs[0];
#pragma unroll
    for (int ee = 0; ee < E_NUM; ++ee) {
        int o1 = offs[ee + 1];
        int seg = o1 - o0;
        int nt = (seg + MTILE - 1) >> 6;
        if (t < nt) {
            e = ee;
            tileStart = o0 + (t << 6);
            rowsLeft = seg - (t << 6);
            if (rowsLeft > MTILE) rowsLeft = MTILE;
            return true;
        }
        t -= nt; o0 = o1;
    }
    return false;
}

// ---------------- stage 1: h = relu(Xg @ W1[e] + b1[e]) -> bf16 hbuf ----------------
__global__ __launch_bounds__(256) void k_ffn1(const float* __restrict__ x,
                                              const float* __restrict__ W1,
                                              const float* __restrict__ b1,
                                              const int* __restrict__ offs,
                                              const int* __restrict__ perm,
                                              unsigned short* __restrict__ hbuf) {
    __shared__ unsigned short lA[MTILE][LDA];
    __shared__ int lPerm[MTILE];
    int tid = threadIdx.x;
    int e, tileStart, rowsLeft;
    if (!tileMap(offs, blockIdx.y, e, tileStart, rowsLeft)) return;
    if (tid < MTILE) lPerm[tid] = (tid < rowsLeft) ? perm[tileStart + tid] : -2;
    __syncthreads();

    int wave = tid >> 5, lane = tid & 31;
    int nlo = lane & 15, khalf = lane >> 4;
    int col = blockIdx.x * 128 + wave * 16 + nlo;      // this wave's N column
    const float* W1e = W1 + (size_t)e * D_DIM * H_DIM;

    v8f acc[4] = {};
    for (int kc = 0; kc < D_DIM; kc += KC) {
        __syncthreads();
        // gather 64 token rows (f32x4 -> bf16x4) into LDS, coalesced along columns
        for (int i = tid; i < MTILE * (KC / 4); i += 256) {
            int r = i >> 6, c4 = i & 63;
            int tok = lPerm[r];
            uint2 pk = make_uint2(0u, 0u);
            if (tok >= 0) {
                float4 v = *(const float4*)&x[(size_t)tok * D_DIM + kc + c4 * 4];
                pk.x = PK2BF(v.x, v.y);
                pk.y = PK2BF(v.z, v.w);
            }
            *(uint2*)&lA[r][c4 * 4] = pk;
        }
        __syncthreads();
        if (kc + KC < D_DIM)
            __builtin_prefetch(W1e + (size_t)(kc + KC + khalf * 16) * H_DIM + col, 0, 1);
        for (int k0 = 0; k0 < KC; k0 += 32) {
            FragBF bf;  // B: 32x16, lane=col, K striped over vector elems (ISA 7.12.2)
            const float* bp = W1e + (size_t)(kc + k0 + khalf * 16) * H_DIM + col;
#pragma unroll
            for (int i = 0; i < 8; ++i)
                bf.u[i] = PK2BF(bp[(size_t)(2 * i) * H_DIM], bp[(size_t)(2 * i + 1) * H_DIM]);
#pragma unroll
            for (int mt = 0; mt < 4; ++mt) {
                FragBF af;  // A: 16x32, lane=row, two 16B groups per lane
                af.q[0] = *(const uint4*)&lA[mt * 16 + nlo][k0 + khalf * 16];
                af.q[1] = *(const uint4*)&lA[mt * 16 + nlo][k0 + khalf * 16 + 8];
                acc[mt] = __builtin_amdgcn_wmma_f32_16x16x32_bf16(
                    false, af.v, false, bf.v, (short)0, acc[mt], false, false);
            }
        }
    }
    float bias = b1[(size_t)e * H_DIM + col];
    if (rowsLeft == MTILE) {            // fast path: full tile, no guards
#pragma unroll
        for (int mt = 0; mt < 4; ++mt) {
#pragma unroll
            for (int v = 0; v < 8; ++v) {
                int r = mt * 16 + khalf * 8 + v;
                float val = acc[mt][v] + bias;
                val = val > 0.f ? val : 0.f;
                hbuf[(size_t)(tileStart + r) * H_DIM + col] = f2bf_r(val);
            }
        }
    } else {
#pragma unroll
        for (int mt = 0; mt < 4; ++mt) {
#pragma unroll
            for (int v = 0; v < 8; ++v) {
                int r = mt * 16 + khalf * 8 + v;
                if (r < rowsLeft) {
                    float val = acc[mt][v] + bias;
                    val = val > 0.f ? val : 0.f;
                    hbuf[(size_t)(tileStart + r) * H_DIM + col] = f2bf_r(val);
                }
            }
        }
    }
}

// ---------------- stage 2: out = h @ W2[e] + b2[e], scatter; sentinel -> Cbuf ----------------
__global__ __launch_bounds__(256) void k_ffn2(const unsigned short* __restrict__ hbuf,
                                              const float* __restrict__ W2,
                                              const float* __restrict__ b2,
                                              const int* __restrict__ offs,
                                              const int* __restrict__ perm,
                                              float* __restrict__ out,
                                              float* __restrict__ Cbuf) {
    __shared__ unsigned short lA[MTILE][LDA];
    __shared__ int lPerm[MTILE];
    int tid = threadIdx.x;
    int e, tileStart, rowsLeft;
    if (!tileMap(offs, blockIdx.y, e, tileStart, rowsLeft)) return;
    if (tid < MTILE) lPerm[tid] = (tid < rowsLeft) ? perm[tileStart + tid] : -2;
    __syncthreads();

    int wave = tid >> 5, lane = tid & 31;
    int nlo = lane & 15, khalf = lane >> 4;
    int col = blockIdx.x * 128 + wave * 16 + nlo;
    const float* W2e = W2 + (size_t)e * H_DIM * D_DIM;

    v8f acc[4] = {};
    for (int kc = 0; kc < H_DIM; kc += KC) {
        __syncthreads();
        // rows already permuted & bf16: straight 16B copies (8 elems each)
#if USE_ASYNC
        for (int i = tid; i < MTILE * (KC / 8); i += 256) {
            int r = i >> 5, c8 = i & 31;
            const unsigned short* src = &hbuf[(size_t)(tileStart + r) * H_DIM + kc + c8 * 8];
            __builtin_amdgcn_global_load_async_to_lds_b128(
                (gv4i_p)src, (lv4i_p)&lA[r][c8 * 8], 0, 0);
        }
        __builtin_amdgcn_s_wait_asynccnt(0);
#else
        for (int i = tid; i < MTILE * (KC / 8); i += 256) {
            int r = i >> 5, c8 = i & 31;
            uint4 v = *(const uint4*)&hbuf[(size_t)(tileStart + r) * H_DIM + kc + c8 * 8];
            *(uint4*)&lA[r][c8 * 8] = v;
        }
#endif
        __syncthreads();
        if (kc + KC < H_DIM)
            __builtin_prefetch(W2e + (size_t)(kc + KC + khalf * 16) * D_DIM + col, 0, 1);
        for (int k0 = 0; k0 < KC; k0 += 32) {
            FragBF bf;
            const float* bp = W2e + (size_t)(kc + k0 + khalf * 16) * D_DIM + col;
#pragma unroll
            for (int i = 0; i < 8; ++i)
                bf.u[i] = PK2BF(bp[(size_t)(2 * i) * D_DIM], bp[(size_t)(2 * i + 1) * D_DIM]);
#pragma unroll
            for (int mt = 0; mt < 4; ++mt) {
                FragBF af;
                af.q[0] = *(const uint4*)&lA[mt * 16 + nlo][k0 + khalf * 16];
                af.q[1] = *(const uint4*)&lA[mt * 16 + nlo][k0 + khalf * 16 + 8];
                acc[mt] = __builtin_amdgcn_wmma_f32_16x16x32_bf16(
                    false, af.v, false, bf.v, (short)0, acc[mt], false, false);
            }
        }
    }
    float bias = b2[(size_t)e * D_DIM + col];
    if (rowsLeft == MTILE && lPerm[0] >= 0) {   // fast path: full tile, no sentinel
#pragma unroll
        for (int mt = 0; mt < 4; ++mt) {
#pragma unroll
            for (int v = 0; v < 8; ++v) {
                int r = mt * 16 + khalf * 8 + v;
                int tok = lPerm[r];
                out[(size_t)tok * D_DIM + col] = acc[mt][v] + bias;
            }
        }
    } else {
#pragma unroll
        for (int mt = 0; mt < 4; ++mt) {
#pragma unroll
            for (int v = 0; v < 8; ++v) {
                int r = mt * 16 + khalf * 8 + v;
                int tok = lPerm[r];
                float val = acc[mt][v] + bias;
                if (tok >= 0)       out[(size_t)tok * D_DIM + col] = val;
                else if (tok == -1) Cbuf[(size_t)e * D_DIM + col] = val;   // C_e row
            }
        }
    }
}

// corr[e][c] = sum_e' C_e'[c] - C_e[c]
__global__ __launch_bounds__(256) void k_corr(const float* __restrict__ Cbuf,
                                              float* __restrict__ corr) {
    int c = blockIdx.x * 256 + threadIdx.x;
    float s = 0.f;
#pragma unroll
    for (int e = 0; e < E_NUM; ++e) s += Cbuf[e * D_DIM + c];
#pragma unroll
    for (int e = 0; e < E_NUM; ++e) corr[e * D_DIM + c] = s - Cbuf[e * D_DIM + c];
}

__global__ __launch_bounds__(256) void k_addcorr(const int* __restrict__ idx,
                                                 const float* __restrict__ corr,
                                                 float* __restrict__ out) {
    int t = blockIdx.x;
    const float* cr = corr + (size_t)idx[t] * D_DIM;
    float* o = out + (size_t)t * D_DIM;
    for (int c = threadIdx.x; c < D_DIM; c += 256) o[c] += cr[c];
}

extern "C" void kernel_launch(void* const* d_in, const int* in_sizes, int n_in,
                              void* d_out, int out_size, void* d_ws, size_t ws_size,
                              hipStream_t stream) {
    const float* x  = (const float*)d_in[0];
    const float* Wg = (const float*)d_in[1];
    const float* bg = (const float*)d_in[2];
    const float* W1 = (const float*)d_in[3];
    const float* b1 = (const float*)d_in[4];
    const float* W2 = (const float*)d_in[5];
    const float* b2 = (const float*)d_in[6];

    float* outMain  = (float*)d_out;                       // [8192,1024] f32
    float* outProbs = outMain + (size_t)NTOK * D_DIM;      // [8192,8]    f32
    int*   outIdx   = (int*)(outProbs + (size_t)NTOK * E_NUM); // [8192] i32

    char* ws = (char*)d_ws;
    int*   counts = (int*)(ws + 0);
    int*   offs   = (int*)(ws + 64);
    int*   cursor = (int*)(ws + 128);
    int*   perm   = (int*)(ws + 256);        // 8208 ints
    float* Cbuf   = (float*)(ws + 33280);    // [8,1024]
    float* corr   = (float*)(ws + 66048);    // [8,1024]
    unsigned short* hbuf = (unsigned short*)(ws + 98816);  // ~67.7 MB bf16 [8264,4096]

    k_init<<<1, 256, 0, stream>>>(counts);
    k_gate<<<NTOK / 8, 256, 0, stream>>>(x, Wg, bg, outProbs, outIdx, counts);
    k_scan<<<1, 32, 0, stream>>>(counts, offs, cursor, perm);
    k_scatter<<<NTOK / 256, 256, 0, stream>>>(outIdx, cursor, perm);

    dim3 g1(H_DIM / 128, MAXTILES);
    k_ffn1<<<g1, 256, 0, stream>>>(x, W1, b1, offs, perm, hbuf);
    dim3 g2(D_DIM / 128, MAXTILES);
    k_ffn2<<<g2, 256, 0, stream>>>(hbuf, W2, b2, offs, perm, outMain, Cbuf);

    k_corr<<<D_DIM / 256, 256, 0, stream>>>(Cbuf, corr);
    k_addcorr<<<NTOK, 256, 0, stream>>>(outIdx, corr, outMain);
}
